// FlashRWLargeAttention_33904471834734
// MI455X (gfx1250) — matrix operand order repaired and checked by reading, not compiled, not run on hardware
//
#include <hip/hip_runtime.h>

// ---------------- types ----------------
typedef __attribute__((ext_vector_type(16))) __bf16 v16bf;
typedef __attribute__((ext_vector_type(8)))  float  v8f;
typedef unsigned int u32;
typedef unsigned short u16;
typedef __attribute__((ext_vector_type(4))) u32 v4u;   // 128-bit chunk (8 bf16)
typedef __attribute__((ext_vector_type(8))) u32 v8u;   // 256-bit (16 bf16)

__device__ __forceinline__ u16 f2bf(float f) {
    u32 u = __builtin_bit_cast(u32, f);
    u32 r = u + 0x7FFFu + ((u >> 16) & 1u);   // round-to-nearest-even
    return (u16)(r >> 16);
}

// Build a v16bf WMMA fragment from two 16-byte chunks.
__device__ __forceinline__ v16bf make_frag(const u16* lo_p, const u16* hi_p) {
    v4u lo = *(const v4u*)lo_p;
    v4u hi = *(const v4u*)hi_p;
    v8u t;
    t[0] = lo[0]; t[1] = lo[1]; t[2] = lo[2]; t[3] = lo[3];
    t[4] = hi[0]; t[5] = hi[1]; t[6] = hi[2]; t[7] = hi[3];
    return __builtin_bit_cast(v16bf, t);
}

#define WMMA_BF16(A, B, C) \
    __builtin_amdgcn_wmma_f32_16x16x32_bf16(false, (A), false, (B), (short)0, (C), false, false)

// Async DMA one 16B chunk global -> LDS (gfx1250, tracked by ASYNCcnt).
// lds_off = low 32 bits of the generic shared pointer (LDS byte offset).
__device__ __forceinline__ void async_copy16(u32 lds_off, const void* gptr) {
    asm volatile("global_load_async_to_lds_b128 %0, %1, off"
                 :: "v"(lds_off), "v"(gptr) : "memory");
}

// ---------------- problem constants ----------------
constexpr int T_TOK  = 2048;   // BATCH*SEQ
constexpr int HID    = 8192;
constexpr int NQKV   = 9216;   // 8 * 18 * 64
constexpr int SEQL   = 1024;
constexpr int NGRP   = 8;      // kv heads
constexpr int NHQ    = 16;     // q heads per kv head
constexpr int DHEAD  = 64;

// ---------------- kernel 1: fp32 -> bf16 (packed x4) ----------------
__global__ void convert_bf16_k(const float* __restrict__ x, u16* __restrict__ y, long n) {
    long i = ((long)blockIdx.x * blockDim.x + threadIdx.x) * 4;
    if (i + 3 < n) {
        float4 f = *(const float4*)(x + i);
        unsigned long long p =  (unsigned long long)f2bf(f.x)
                             | ((unsigned long long)f2bf(f.y) << 16)
                             | ((unsigned long long)f2bf(f.z) << 32)
                             | ((unsigned long long)f2bf(f.w) << 48);
        *(unsigned long long*)(y + i) = p;
    }
}

// ---------------- kernel 2: transpose + convert W[K][N] fp32 -> Wt[N][K] bf16 ----------------
__global__ void transpose_convert_k(const float* __restrict__ W, u16* __restrict__ Wt,
                                    int K, int N) {
    __shared__ float tile[32][33];
    int nb = blockIdx.x * 32;
    int kb = blockIdx.y * 32;
    for (int i = threadIdx.y; i < 32; i += 8)
        tile[i][threadIdx.x] = W[(long)(kb + i) * N + nb + threadIdx.x];
    __syncthreads();
    for (int i = threadIdx.y; i < 32; i += 8)
        Wt[(long)(nb + i) * K + kb + threadIdx.x] = f2bf(tile[threadIdx.x][i]);
}

// ---------------- kernel 3: bf16 GEMM  C[M][ldc](fp32) = A[M][K] * Bt[N][K]^T ----------------
// Block tile 128(M) x 256(N), BK=32, 8 waves (2 x 4), wave tile 64x64 (4x4 WMMA tiles).
// Double-buffered LDS filled by async global->LDS DMA, one tile of prefetch distance.
__global__ __launch_bounds__(256) void gemm_bf16_k(const u16* __restrict__ A,
                                                   const u16* __restrict__ Bt,
                                                   float* __restrict__ C,
                                                   int M, int N, int K, int ldc) {
    __shared__ __align__(16) u16 As[2][128][32];   // 16 KiB
    __shared__ __align__(16) u16 Bs[2][256][32];   // 32 KiB

    const int tid  = threadIdx.x;
    const int lane = tid & 31;
    const int wave = tid >> 5;
    const int half = lane >> 4;
    const int l16  = lane & 15;
    const int wm   = wave >> 2;   // 0..1
    const int wn   = wave & 3;    // 0..3
    const int m0   = blockIdx.y * 128;
    const int n0   = blockIdx.x * 256;

    // Issue one BK=32 tile (A: 2 chunks/thread, B: 4 chunks/thread) as async DMA.
    auto issue_tile = [&](int k0, int buf) {
        #pragma unroll
        for (int rep = 0; rep < 2; ++rep) {
            int cc = tid + rep * 256;          // 0..511
            int row = cc >> 2, col = (cc & 3) * 8;
            async_copy16((u32)(unsigned long long)&As[buf][row][col],
                         &A[(long)(m0 + row) * K + k0 + col]);
        }
        #pragma unroll
        for (int rep = 0; rep < 4; ++rep) {
            int cc = tid + rep * 256;          // 0..1023
            int row = cc >> 2, col = (cc & 3) * 8;
            async_copy16((u32)(unsigned long long)&Bs[buf][row][col],
                         &Bt[(long)(n0 + row) * K + k0 + col]);
        }
    };

    v8f acc[4][4] = {};
    const int ntiles = K / 32;

    issue_tile(0, 0);
    for (int it = 0; it < ntiles; ++it) {
        const int buf = it & 1;
        if (it + 1 < ntiles) {
            issue_tile((it + 1) * 32, buf ^ 1);
            // 6 ops in flight for tile it+1; <=6 means tile it's 6 are complete
            asm volatile("s_wait_asynccnt 0x6" ::: "memory");
        } else {
            asm volatile("s_wait_asynccnt 0x0" ::: "memory");
        }
        __syncthreads();

        v16bf afrag[4], bfrag[4];
        #pragma unroll
        for (int mi = 0; mi < 4; ++mi) {
            int row = wm * 64 + mi * 16 + l16;
            afrag[mi] = make_frag(&As[buf][row][half * 8], &As[buf][row][16 + half * 8]);
        }
        #pragma unroll
        for (int ni = 0; ni < 4; ++ni) {
            int n = wn * 64 + ni * 16 + l16;
            bfrag[ni] = make_frag(&Bs[buf][n][half * 8], &Bs[buf][n][16 + half * 8]);
        }

        #pragma unroll
        for (int mi = 0; mi < 4; ++mi)
            #pragma unroll
            for (int ni = 0; ni < 4; ++ni)
                acc[mi][ni] = WMMA_BF16(afrag[mi], bfrag[ni], acc[mi][ni]);

        __syncthreads();   // protect buf from next iteration's async writes
    }

    // ---- epilogue: C layout (VGPR r, lanes 0-15 -> M=r, lanes 16-31 -> M=r+8) ----
    #pragma unroll
    for (int mi = 0; mi < 4; ++mi)
        #pragma unroll
        for (int ni = 0; ni < 4; ++ni)
            #pragma unroll
            for (int r = 0; r < 8; ++r) {
                int row = m0 + wm * 64 + mi * 16 + r + 8 * half;
                int col = n0 + wn * 64 + ni * 16 + l16;
                C[(long)row * ldc + col] = acc[mi][ni][r];
            }
}

// ---------------- kernel 4: RoPE + reshape qkv(fp32) -> Q/K/Vt (bf16) ----------------
// Q: [b][g][hq][s][64], K: [b][g][s][64], Vt: [b][g][64][s]
__global__ void qkv_post_k(const float* __restrict__ qkv,
                           const float* __restrict__ cosp,
                           const float* __restrict__ sinp,
                           u16* __restrict__ Qo, u16* __restrict__ Ko, u16* __restrict__ Vto) {
    int t = blockIdx.x;          // token 0..2047
    int g = blockIdx.y;          // kv head
    int b = t >> 10;
    int s = t & 1023;
    const float* src = qkv + (long)t * NQKV + g * ((NHQ + 2) * DHEAD);

    for (int i = threadIdx.x; i < (NHQ + 2) * 32; i += blockDim.x) {
        int head = i >> 5;
        int j    = i & 31;       // rotary pair index
        float x1 = src[head * 64 + j];
        float x2 = src[head * 64 + j + 32];
        if (head < NHQ + 1) {
            float c  = cosp[(long)t * 32 + j];
            float sn = sinp[(long)t * 32 + j];
            float y1 = x1 * c - x2 * sn;
            float y2 = x2 * c + x1 * sn;
            if (head < NHQ) {
                u16* q = Qo + ((((long)(b * NGRP + g) * NHQ + head) * SEQL + s) * DHEAD);
                q[j]      = f2bf(y1);
                q[j + 32] = f2bf(y2);
            } else {
                u16* k = Ko + (((long)(b * NGRP + g) * SEQL + s) * DHEAD);
                k[j]      = f2bf(y1);
                k[j + 32] = f2bf(y2);
            }
        } else {
            u16* v = Vto + ((long)(b * NGRP + g) * DHEAD) * SEQL;
            v[(long)j * SEQL + s]        = f2bf(x1);
            v[(long)(j + 32) * SEQL + s] = f2bf(x2);
        }
    }
}

// ---------------- kernel 5: causal flash attention (per-wave 16 query rows) ----------------
// grid.x = b(2) * g(8) * hq(16) * qtile(8); block = 256 = 8 waves; wave handles 16 queries.
__global__ __launch_bounds__(256) void attn_k(const u16* __restrict__ Q,
                                              const u16* __restrict__ Kc,
                                              const u16* __restrict__ Vt,
                                              u16* __restrict__ Out) {
    int bid = blockIdx.x;
    int qt  = bid & 7;  bid >>= 3;
    int hq  = bid & 15; bid >>= 4;
    int g   = bid & 7;  bid >>= 3;
    int b   = bid;

    const int lane = threadIdx.x & 31;
    const int wave = threadIdx.x >> 5;
    const int half = lane >> 4;
    const int l16  = lane & 15;
    const int q0   = qt * 128 + wave * 16;

    const u16* Qp = Q  + (((long)(b * NGRP + g) * NHQ + hq) * SEQL) * DHEAD;
    const u16* Kp = Kc + ((long)(b * NGRP + g) * SEQL) * DHEAD;
    const u16* Vp = Vt + ((long)(b * NGRP + g) * DHEAD) * SEQL;

    __shared__ __align__(16) u16 p_lds[8][16][32];   // per-wave private P tile

    // Q fragments: two k-steps over d (0..31, 32..63)
    v16bf qf[2];
    #pragma unroll
    for (int d = 0; d < 2; ++d) {
        const u16* base = &Qp[(long)(q0 + l16) * DHEAD + d * 32];
        qf[d] = make_frag(base + half * 8, base + 16 + half * 8);
    }

    float mrow[8], lrow[8];
    #pragma unroll
    for (int r = 0; r < 8; ++r) { mrow[r] = -1e30f; lrow[r] = 0.0f; }
    v8f o[4] = {};   // output d-slices n*16

    const int kv_end = q0 + 16;            // causal: kv <= q0+15
    for (int kv0 = 0; kv0 < kv_end; kv0 += 32) {
        // ---- S = Q * K^T  (two 16-wide kv sub-tiles, reduce over d=64) ----
        v8f s[2] = {};
        #pragma unroll
        for (int nf = 0; nf < 2; ++nf) {
            const u16* kb = &Kp[(long)(kv0 + nf * 16 + l16) * DHEAD];
            #pragma unroll
            for (int d = 0; d < 2; ++d) {
                v16bf kf = make_frag(kb + d * 32 + half * 8, kb + d * 32 + 16 + half * 8);
                s[nf] = WMMA_BF16(qf[d], kf, s[nf]);
            }
        }

        // ---- scale + causal mask ----
        #pragma unroll
        for (int nf = 0; nf < 2; ++nf) {
            int kv = kv0 + nf * 16 + l16;
            #pragma unroll
            for (int r = 0; r < 8; ++r) {
                int qrow = q0 + r + 8 * half;
                float v  = s[nf][r] * 0.125f;          // 64^-0.5
                s[nf][r] = (kv <= qrow) ? v : -3.0e38f;
            }
        }

        // ---- online softmax (row lives in 16 lanes of one half) ----
        float alpha[8];
        #pragma unroll
        for (int r = 0; r < 8; ++r) {
            float rm = fmaxf(s[0][r], s[1][r]);
            #pragma unroll
            for (int msk = 1; msk <= 8; msk <<= 1)
                rm = fmaxf(rm, __shfl_xor(rm, msk, 32));
            float mnew = fmaxf(mrow[r], rm);
            alpha[r] = __expf(mrow[r] - mnew);
            mrow[r]  = mnew;
        }

        float rsum[8];
        #pragma unroll
        for (int r = 0; r < 8; ++r) rsum[r] = 0.0f;
        #pragma unroll
        for (int nf = 0; nf < 2; ++nf)
            #pragma unroll
            for (int r = 0; r < 8; ++r) {
                float p = __expf(s[nf][r] - mrow[r]);
                rsum[r] += p;
                p_lds[wave][r + 8 * half][nf * 16 + l16] = f2bf(p);
            }
        #pragma unroll
        for (int r = 0; r < 8; ++r) {
            float rs = rsum[r];
            #pragma unroll
            for (int msk = 1; msk <= 8; msk <<= 1)
                rs += __shfl_xor(rs, msk, 32);
            lrow[r] = lrow[r] * alpha[r] + rs;
        }

        // per-wave LDS transpose: wait for our own ds stores to land
        asm volatile("s_wait_dscnt 0x0" ::: "memory");

        // P as A-fragment (16x32)
        v16bf pf = make_frag(&p_lds[wave][l16][half * 8],
                             &p_lds[wave][l16][16 + half * 8]);

        // ---- O = O*alpha + P * V  (4 d-slices of 16) ----
        #pragma unroll
        for (int n = 0; n < 4; ++n) {
            const u16* vb = &Vp[(long)(n * 16 + l16) * SEQL + kv0];
            v16bf vf = make_frag(vb + half * 8, vb + 16 + half * 8);
            #pragma unroll
            for (int r = 0; r < 8; ++r) o[n][r] *= alpha[r];
            o[n] = WMMA_BF16(pf, vf, o[n]);
        }
    }

    // ---- epilogue: normalize and write bf16 attn output [T][8192] ----
    #pragma unroll
    for (int n = 0; n < 4; ++n)
        #pragma unroll
        for (int r = 0; r < 8; ++r) {
            int qrow = q0 + r + 8 * half;
            int tok  = b * SEQL + qrow;
            float val = o[n][r] / lrow[r];
            Out[(long)tok * HID + (g * NHQ + hq) * DHEAD + n * 16 + l16] = f2bf(val);
        }
}

// ---------------- launcher ----------------
extern "C" void kernel_launch(void* const* d_in, const int* in_sizes, int n_in,
                              void* d_out, int out_size, void* d_ws, size_t ws_size,
                              hipStream_t stream) {
    const float* hs      = (const float*)d_in[0];   // (2048, 8192)
    const float* cosp    = (const float*)d_in[1];   // (2048, 32)
    const float* sinp    = (const float*)d_in[2];   // (2048, 32)
    const float* W_qkv   = (const float*)d_in[3];   // (8192, 9216)
    const float* W_dense = (const float*)d_in[4];   // (8192, 8192)
    float* out = (float*)d_out;                     // (2048, 8192) fp32

    char* ws = (char*)d_ws;
    // static workspace layout (all sizes 256B-aligned by construction)
    u16*   hs_bf   = (u16*)  (ws + 0);           // 2048*8192*2   =  32 MiB
    u16*   wqkvt   = (u16*)  (ws + 33554432LL);  // 9216*8192*2   = 144 MiB
    u16*   wdt     = (u16*)  (ws + 184549376LL); // 8192*8192*2   = 128 MiB
    float* qkvf    = (float*)(ws + 318767104LL); // 2048*9216*4   =  72 MiB
    u16*   q_bf    = (u16*)  (ws + 394264576LL); // 2*8*16*1024*64*2
    u16*   k_bf    = (u16*)  (ws + 427819008LL); // 2*8*1024*64*2
    u16*   vt_bf   = (u16*)  (ws + 429916160LL); // 2*8*64*1024*2
    u16*   attn_bf = (u16*)  (ws + 432013312LL); // 2048*8192*2

    // 1) hidden_states -> bf16
    {
        long n = (long)T_TOK * HID;
        convert_bf16_k<<<(unsigned)(n / (256 * 4)), 256, 0, stream>>>(hs, hs_bf, n);
    }
    // 2) weights -> bf16, transposed (reduction-major); bf16 copies fit in 192MB L2
    transpose_convert_k<<<dim3(NQKV / 32, HID / 32), dim3(32, 8), 0, stream>>>(W_qkv, wqkvt, HID, NQKV);
    transpose_convert_k<<<dim3(HID  / 32, HID / 32), dim3(32, 8), 0, stream>>>(W_dense, wdt, HID, HID);

    // 3) QKV GEMM: (2048 x 8192) * (8192 x 9216) -> fp32
    gemm_bf16_k<<<dim3(NQKV / 256, T_TOK / 128), 256, 0, stream>>>(
        hs_bf, wqkvt, qkvf, T_TOK, NQKV, HID, NQKV);

    // 4) RoPE + reshape to attention layouts (V transposed for WMMA B-fragments)
    qkv_post_k<<<dim3(T_TOK, NGRP), 256, 0, stream>>>(qkvf, cosp, sinp, q_bf, k_bf, vt_bf);

    // 5) causal GQA flash attention
    attn_k<<<dim3(2 * NGRP * NHQ * (SEQL / 128)), 256, 0, stream>>>(q_bf, k_bf, vt_bf, attn_bf);

    // 6) dense GEMM: (2048 x 8192) * (8192 x 8192) -> fp32 d_out
    gemm_bf16_k<<<dim3(HID / 256, T_TOK / 128), 256, 0, stream>>>(
        attn_bf, wdt, out, T_TOK, HID, HID, HID);

    (void)in_sizes; (void)n_in; (void)out_size; (void)ws_size;
}
